// StructureModule_60782377173333
// MI455X (gfx1250) — matrix operand order repaired and checked by reading, not compiled, not run on hardware
//
#include <hip/hip_runtime.h>
#include <math.h>

// ---------------------------------------------------------------------------
// Problem constants (B=1)
// ---------------------------------------------------------------------------
#define NN    512
#define CS    384
#define CZ    128
#define NH    12
#define DD    16
#define PQn   4
#define PVn   8
#define PROJC 1152          // 192(q)+192(k)+192(v)+144(qf)+144(kf)+288(vf)
#define CATC  2112          // 192 rep + 288 vals + 96 norms + 1536 pair
#define COL_Q   0
#define COL_K   192
#define COL_V   384
#define COL_QF  576
#define COL_KF  720
#define COL_VF  864

__device__ __constant__ const float kWC = 0.2357022603955158f;   // sqrt(2/(9*4))
__device__ __constant__ const float kWL = 0.5773502691896258f;   // sqrt(1/3)

typedef __attribute__((ext_vector_type(8)))  __bf16 v8bf;
typedef __attribute__((ext_vector_type(16))) __bf16 v16bf;
typedef __attribute__((ext_vector_type(8)))  float  v8f;

// ---------------------------------------------------------------------------
// CDNA5 async global->LDS path (ASYNCcnt), guarded so absence falls back to a
// synchronous float4 copy. Builtin signature (from hipcc diagnostics):
//   void __builtin_amdgcn_global_load_async_to_lds_b128(
//       int __attribute__((vector_size(16))) AS1* src,
//       int __attribute__((vector_size(16))) AS3* dst, int imm_off, int cpol)
// ---------------------------------------------------------------------------
#if defined(__has_builtin)
#if __has_builtin(__builtin_amdgcn_global_load_async_to_lds_b128)
#define HAVE_ASYNC_LDS 1
#endif
#endif

#if defined(HAVE_ASYNC_LDS)
typedef int v4i_vs __attribute__((vector_size(4 * sizeof(int))));
typedef __attribute__((address_space(1))) v4i_vs gas_v4i;
typedef __attribute__((address_space(3))) v4i_vs las_v4i;
// generic->AS1: full 64-bit address; generic->AS3: low 32 bits are the LDS
// byte offset (ISA 10.2 aperture mapping), so integer truncation is exact.
#define ASYNC_CP_B128(gsrc, ldst)                                              \
  __builtin_amdgcn_global_load_async_to_lds_b128(                              \
      (gas_v4i*)(uintptr_t)(gsrc), (las_v4i*)(uint32_t)(uintptr_t)(ldst), 0, 0)
#if __has_builtin(__builtin_amdgcn_s_wait_asynccnt)
#define WAIT_ASYNC(n) __builtin_amdgcn_s_wait_asynccnt(n)
#else
#define WAIT_ASYNC(n) asm volatile("s_wait_asynccnt %0" ::"i"(n))
#endif
#endif

// ---------------------------------------------------------------------------
// bf16 WMMA fragment loaders (ISA 7.12.2 layouts, wave32)
// A: 16x32 row-major, lane(l): row = l&15; lanes 0-15 hold K {0..7,16..23},
//    lanes 16-31 hold K {8..15,24..31}.
// B: 32x16 with B stored K-contiguous (we pre-transpose weights): lane(l):
//    col = l&15; lanes 0-15 hold K {0..7,16..23}, lanes 16-31 hold K {8..15,24..31}.
// ---------------------------------------------------------------------------
__device__ inline v16bf load_a_frag(const __bf16* __restrict__ A, int lda,
                                    int m0, int k0, int lane) {
  int r  = lane & 15;
  int hi = lane >> 4;
  const __bf16* p = A + (size_t)(m0 + r) * lda + (k0 + hi * 8);
  v8bf lo = *(const v8bf*)(p);
  v8bf hh = *(const v8bf*)(p + 16);
  return __builtin_shufflevector(lo, hh, 0,1,2,3,4,5,6,7,8,9,10,11,12,13,14,15);
}

__device__ inline v16bf load_b_frag(const __bf16* __restrict__ Bt, int ldb,
                                    int n0, int k0, int lane) {
  int n  = lane & 15;
  int hi = lane >> 4;
  const __bf16* p = Bt + (size_t)(n0 + n) * ldb + (k0 + hi * 8);
  v8bf lo = *(const v8bf*)(p);
  v8bf hh = *(const v8bf*)(p + 16);
  return __builtin_shufflevector(lo, hh, 0,1,2,3,4,5,6,7,8,9,10,11,12,13,14,15);
}

// ---------------------------------------------------------------------------
// bf16 WMMA GEMM, 16x64 tile per wave: A-fragment reused across 4 WMMAs.
// C[M,N] (f32) = A[M,K](bf16 rm) x Bt[N,K](bf16, pre-transposed) + bias.
// grid = (N/64, M/16), block = 32 (one wave).
// ---------------------------------------------------------------------------
__global__ __launch_bounds__(32)
void k_gemm_bf16(const __bf16* __restrict__ A, const __bf16* __restrict__ Bt,
                 float* __restrict__ C, int M, int N, int K,
                 const float* __restrict__ bias) {
  const int n0   = blockIdx.x * 64;
  const int m0   = blockIdx.y * 16;
  const int lane = threadIdx.x;

  v8f c0 = {}, c1 = {}, c2 = {}, c3 = {};
  for (int k0 = 0; k0 < K; k0 += 32) {
    v16bf a  = load_a_frag(A, K, m0, k0, lane);
    v16bf b0 = load_b_frag(Bt, K, n0 +  0, k0, lane);
    v16bf b1 = load_b_frag(Bt, K, n0 + 16, k0, lane);
    v16bf b2 = load_b_frag(Bt, K, n0 + 32, k0, lane);
    v16bf b3 = load_b_frag(Bt, K, n0 + 48, k0, lane);
    c0 = __builtin_amdgcn_wmma_f32_16x16x32_bf16(false, a, false, b0, (short)0, c0, false, false);
    c1 = __builtin_amdgcn_wmma_f32_16x16x32_bf16(false, a, false, b1, (short)0, c1, false, false);
    c2 = __builtin_amdgcn_wmma_f32_16x16x32_bf16(false, a, false, b2, (short)0, c2, false, false);
    c3 = __builtin_amdgcn_wmma_f32_16x16x32_bf16(false, a, false, b3, (short)0, c3, false, false);
  }
  // D layout: element r -> row m0 + r + 8*(lane>>4), col ntile + (lane&15)
  const int rbase = m0 + ((lane >> 4) << 3);
  v8f acc[4] = {c0, c1, c2, c3};
#pragma unroll
  for (int nt = 0; nt < 4; ++nt) {
    const int col  = n0 + nt * 16 + (lane & 15);
    const float bv = bias ? bias[col] : 0.0f;
#pragma unroll
    for (int r = 0; r < 8; ++r)
      C[(size_t)(rbase + r) * N + col] = acc[nt][r] + bv;
  }
}

// ---------------------------------------------------------------------------
// Pack: s -> bf16, projection weights -> concatenated+transposed bf16,
//       w_out -> transposed bf16
// ---------------------------------------------------------------------------
__global__ void k_pack(const float* __restrict__ s,
                       const float* __restrict__ wq,  const float* __restrict__ wk,
                       const float* __restrict__ wv,  const float* __restrict__ wqf,
                       const float* __restrict__ wkf, const float* __restrict__ wvf,
                       const float* __restrict__ wout,
                       __bf16* __restrict__ sb, __bf16* __restrict__ wcat_t,
                       __bf16* __restrict__ wout_t) {
  int idx = blockIdx.x * 256 + threadIdx.x;
  const int NS = NN * CS;            // 196608
  const int NW = PROJC * CS;         // 442368
  const int NO = CS * CATC;          // 811008
  if (idx < NS) { sb[idx] = (__bf16)s[idx]; return; }
  idx -= NS;
  if (idx < NW) {                    // wcat_t[c][r] = Wcat[r][c]
    int c = idx / CS, r = idx - c * CS;
    float v;
    if      (c < 192) v = wq [r * 192 + c];
    else if (c < 384) v = wk [r * 192 + (c - 192)];
    else if (c < 576) v = wv [r * 192 + (c - 384)];
    else if (c < 720) v = wqf[r * 144 + (c - 576)];
    else if (c < 864) v = wkf[r * 144 + (c - 720)];
    else              v = wvf[r * 288 + (c - 864)];
    wcat_t[idx] = (__bf16)v; return;
  }
  idx -= NW;
  if (idx < NO) {                    // wout_t[n][k] = w_out[k][n]
    int n = idx / CATC, k = idx - n * CATC;
    wout_t[idx] = (__bf16)wout[(size_t)k * CS + n];
  }
}

// ---------------------------------------------------------------------------
// Lift local-frame points to global frame: g = R*p + t
// grid = 512 blocks, 192 threads: t<48 -> gq, t<96 -> gk, else gv
// ---------------------------------------------------------------------------
__global__ __launch_bounds__(192)
void k_frames(const float* __restrict__ proj, const float* __restrict__ rot,
              const float* __restrict__ trans,
              float* __restrict__ gq, float* __restrict__ gk,
              float* __restrict__ gv) {
  const int i = blockIdx.x;
  const int t = threadIdx.x;
  const float* R = rot + (size_t)i * 9;
  const float* T = trans + (size_t)i * 3;
  const float* prow = proj + (size_t)i * PROJC;

  const float* src;
  float* dst;
  if (t < 48)      { src = prow + COL_QF + t * 3;        dst = gq + (size_t)i * 144 + t * 3; }
  else if (t < 96) { src = prow + COL_KF + (t - 48) * 3; dst = gk + (size_t)i * 144 + (t - 48) * 3; }
  else             { src = prow + COL_VF + (t - 96) * 3; dst = gv + (size_t)i * 288 + (t - 96) * 3; }

  const float x = src[0], y = src[1], z = src[2];
#pragma unroll
  for (int o = 0; o < 3; ++o)
    dst[o] = R[o * 3 + 0] * x + R[o * 3 + 1] * y + R[o * 3 + 2] * z + T[o];
}

// ---------------------------------------------------------------------------
// Attention core: one workgroup per query i. z[i,:,:] (256 KB) is pulled once
// for bias/scores and re-read from L2 for out_pair (192 MB L2 keeps it hot).
// out_pair staging uses double-buffered async global->LDS copies (ASYNCcnt).
// ---------------------------------------------------------------------------
#define TJ 16   // z tile rows per stage (16*128 floats = 8 KB per buffer)

__global__ __launch_bounds__(256)
void k_attn(const float* __restrict__ z,    const float* __restrict__ proj,
            const float* __restrict__ gq,   const float* __restrict__ gk,
            const float* __restrict__ gv,   const float* __restrict__ rot,
            const float* __restrict__ trans,const float* __restrict__ w_bias,
            const float* __restrict__ head_w, __bf16* __restrict__ catb) {
  __shared__ float sc[NN * NH];         // scores -> attn weights   24576 B
  __shared__ float zt[2][TJ * CZ];      // double-buffered z tile   16384 B
  __shared__ float wb[CZ * NH];         // bias weights              6144 B
  __shared__ float Qi[NH * DD];         // query row                  768 B
  __shared__ float gqi[NH * PQn * 3];   // global query points        576 B
  __shared__ float gam[NH];
  __shared__ float red[256];
  __shared__ float tmps[NH * PVn * 3];

  const int i   = blockIdx.x;
  const int tid = threadIdx.x;

  // ---- setup ----
  if (tid < NH * DD)       Qi[tid]  = proj[(size_t)i * PROJC + COL_Q + tid];
  if (tid < NH * PQn * 3)  gqi[tid] = gq[(size_t)i * 144 + tid];
  if (tid < NH)            gam[tid] = log1pf(__expf(head_w[tid]));  // softplus
  for (int idx = tid; idx < CZ * NH; idx += 256) wb[idx] = w_bias[idx];
  __syncthreads();

  // ---- phase B: scores[j][h] ----
  for (int j = tid; j < NN; j += 256) {
    float bh[NH], qk[NH], pt[NH];
#pragma unroll
    for (int h = 0; h < NH; ++h) { bh[h] = 0.f; qk[h] = 0.f; pt[h] = 0.f; }

    const float4* z4 = (const float4*)(z + ((size_t)i * NN + j) * CZ);
    __builtin_prefetch(z + ((size_t)i * NN + (j + 256 < NN ? j + 256 : j)) * CZ, 0, 0);
    for (int d4 = 0; d4 < CZ / 4; ++d4) {
      const float4 zv = z4[d4];
      const int d = d4 * 4;
#pragma unroll
      for (int h = 0; h < NH; ++h)
        bh[h] += zv.x * wb[(d + 0) * NH + h] + zv.y * wb[(d + 1) * NH + h] +
                 zv.z * wb[(d + 2) * NH + h] + zv.w * wb[(d + 3) * NH + h];
    }
    const float* krow = proj + (size_t)j * PROJC + COL_K;
#pragma unroll
    for (int h = 0; h < NH; ++h)
      for (int d = 0; d < DD; ++d) qk[h] += Qi[h * DD + d] * krow[h * DD + d];

    const float* gkr = gk + (size_t)j * 144;
#pragma unroll
    for (int h = 0; h < NH; ++h)
      for (int e = 0; e < PQn * 3; ++e) {
        float df = gqi[h * 12 + e] - gkr[h * 12 + e];
        pt[h] += df * df;
      }
#pragma unroll
    for (int h = 0; h < NH; ++h)
      sc[j * NH + h] = kWL * (qk[h] * 0.25f + bh[h] - 0.5f * kWC * gam[h] * pt[h]);
  }
  __syncthreads();

  // ---- phase C: softmax over j, per head ----
  for (int h = 0; h < NH; ++h) {
    float m = -1e30f;
    for (int j = tid; j < NN; j += 256) m = fmaxf(m, sc[j * NH + h]);
    red[tid] = m; __syncthreads();
    for (int s = 128; s > 0; s >>= 1) {
      if (tid < s) red[tid] = fmaxf(red[tid], red[tid + s]);
      __syncthreads();
    }
    m = red[0]; __syncthreads();

    float ss = 0.f;
    for (int j = tid; j < NN; j += 256) {
      float e = __expf(sc[j * NH + h] - m);
      sc[j * NH + h] = e;
      ss += e;
    }
    red[tid] = ss; __syncthreads();
    for (int s = 128; s > 0; s >>= 1) {
      if (tid < s) red[tid] += red[tid + s];
      __syncthreads();
    }
    float inv = 1.0f / red[0]; __syncthreads();
    for (int j = tid; j < NN; j += 256) sc[j * NH + h] *= inv;
    __syncthreads();
  }

  __bf16* crow = catb + (size_t)i * CATC;

  // ---- phase D1: out_pair[h][d] via double-buffered z tiles ----
  float op[6];
#pragma unroll
  for (int u = 0; u < 6; ++u) op[u] = 0.f;
  const int pbase = tid * 6;             // 256*6 == 1536 == NH*CZ
  const float4* zbase4 = (const float4*)(z + (size_t)i * NN * CZ);
  const int NT = NN / TJ;                // 32 tiles, TJ*CZ/4 = 512 float4/tile

#if defined(HAVE_ASYNC_LDS)
  // prologue: stage tile 0 into buffer 0 (2 async b128 ops per thread)
#pragma unroll
  for (int u = 0; u < 2; ++u) {
    const int idx = tid + u * 256;
    ASYNC_CP_B128(zbase4 + idx, &zt[0][idx * 4]);
  }
  for (int t = 0; t < NT; ++t) {
    const int cur = t & 1;
    if (t + 1 < NT) {
#pragma unroll
      for (int u = 0; u < 2; ++u) {
        const int idx = tid + u * 256;
        ASYNC_CP_B128(zbase4 + (size_t)(t + 1) * (TJ * CZ / 4) + idx,
                      &zt[1 - cur][idx * 4]);
      }
      WAIT_ASYNC(2);   // tile t complete; tile t+1 still in flight
    } else {
      WAIT_ASYNC(0);
    }
    __syncthreads();
    for (int jj = 0; jj < TJ; ++jj) {
#pragma unroll
      for (int u = 0; u < 6; ++u) {
        const int p = pbase + u;
        op[u] += sc[(t * TJ + jj) * NH + (p >> 7)] * zt[cur][jj * CZ + (p & 127)];
      }
    }
    __syncthreads();
  }
#else
  for (int t = 0; t < NT; ++t) {
    __syncthreads();
    float4* zd = (float4*)zt[0];
    for (int idx = tid; idx < TJ * CZ / 4; idx += 256)
      zd[idx] = zbase4[(size_t)t * (TJ * CZ / 4) + idx];
    __syncthreads();
    for (int jj = 0; jj < TJ; ++jj) {
#pragma unroll
      for (int u = 0; u < 6; ++u) {
        const int p = pbase + u;
        op[u] += sc[(t * TJ + jj) * NH + (p >> 7)] * zt[0][jj * CZ + (p & 127)];
      }
    }
  }
#endif
#pragma unroll
  for (int u = 0; u < 6; ++u) crow[576 + pbase + u] = (__bf16)op[u];
  __syncthreads();

  // ---- phase D2: out_rep[h][d] = sum_j attn * V ----
  if (tid < NH * DD) {
    const int h = tid >> 4, d = tid & 15;
    float acc = 0.f;
    for (int j = 0; j < NN; ++j)
      acc += sc[j * NH + h] * proj[(size_t)j * PROJC + COL_V + h * DD + d];
    crow[tid] = (__bf16)acc;
  }

  // ---- phase D3: tmp[h][q][o] = sum_j attn * gv ----
  for (int e = tid; e < NH * PVn * 3; e += 256) {
    const int h = e / 24;
    float acc = 0.f;
    for (int j = 0; j < NN; ++j)
      acc += sc[j * NH + h] * gv[(size_t)j * 288 + e];
    tmps[e] = acc;
  }
  __syncthreads();

  // ---- phase E: inverse frame + norms ----
  if (tid < NH * PVn) {
    const int h = tid >> 3, q = tid & 7;
    const float* R = rot + (size_t)i * 9;
    const float t0 = tmps[h * 24 + q * 3 + 0] - trans[i * 3 + 0];
    const float t1 = tmps[h * 24 + q * 3 + 1] - trans[i * 3 + 1];
    const float t2 = tmps[h * 24 + q * 3 + 2] - trans[i * 3 + 2];
    float ov[3];
#pragma unroll
    for (int o = 0; o < 3; ++o)        // R_inv = R^T
      ov[o] = R[0 * 3 + o] * t0 + R[1 * 3 + o] * t1 + R[2 * 3 + o] * t2;
    const float nrm = sqrtf(ov[0] * ov[0] + ov[1] * ov[1] + ov[2] * ov[2] + 1e-8f);
#pragma unroll
    for (int o = 0; o < 3; ++o) crow[192 + h * 24 + q * 3 + o] = (__bf16)ov[o];
    crow[480 + h * 8 + q] = (__bf16)nrm;
  }
}

// ---------------------------------------------------------------------------
// Host-side launcher
// ---------------------------------------------------------------------------
extern "C" void kernel_launch(void* const* d_in, const int* in_sizes, int n_in,
                              void* d_out, int out_size, void* d_ws, size_t ws_size,
                              hipStream_t stream) {
  const float* s       = (const float*)d_in[0];
  const float* z       = (const float*)d_in[1];
  const float* rot     = (const float*)d_in[2];
  const float* trans   = (const float*)d_in[3];
  const float* w_q     = (const float*)d_in[4];
  const float* w_k     = (const float*)d_in[5];
  const float* w_v     = (const float*)d_in[6];
  const float* w_qf    = (const float*)d_in[7];
  const float* w_kf    = (const float*)d_in[8];
  const float* w_vf    = (const float*)d_in[9];
  const float* w_bias  = (const float*)d_in[10];
  const float* head_w  = (const float*)d_in[11];
  const float* w_out   = (const float*)d_in[12];
  const float* b_out   = (const float*)d_in[13];
  float* out = (float*)d_out;

  char* ws = (char*)d_ws;
  size_t off = 0;
  auto carve = [&](size_t bytes) -> void* {
    void* p = (void*)(ws + off);
    off += (bytes + 255) & ~(size_t)255;
    return p;
  };

  __bf16* sb     = (__bf16*)carve((size_t)NN * CS * sizeof(__bf16));
  __bf16* wcat_t = (__bf16*)carve((size_t)PROJC * CS * sizeof(__bf16));
  __bf16* wout_t = (__bf16*)carve((size_t)CS * CATC * sizeof(__bf16));
  __bf16* catb   = (__bf16*)carve((size_t)NN * CATC * sizeof(__bf16));
  float*  proj   = (float*) carve((size_t)NN * PROJC * sizeof(float));
  float*  gqg    = (float*) carve((size_t)NN * 144 * sizeof(float));
  float*  gkg    = (float*) carve((size_t)NN * 144 * sizeof(float));
  float*  gvg    = (float*) carve((size_t)NN * 288 * sizeof(float));

  // 1) precision pack
  const int packN = NN * CS + PROJC * CS + CS * CATC;
  k_pack<<<(packN + 255) / 256, 256, 0, stream>>>(
      s, w_q, w_k, w_v, w_qf, w_kf, w_vf, w_out, sb, wcat_t, wout_t);

  // 2) fused projection GEMM: proj[512,1152] = s @ [wq|wk|wv|wqf|wkf|wvf]
  k_gemm_bf16<<<dim3(PROJC / 64, NN / 16), 32, 0, stream>>>(
      sb, wcat_t, proj, NN, PROJC, CS, nullptr);

  // 3) lift points to global frames
  k_frames<<<NN, 192, 0, stream>>>(proj, rot, trans, gqg, gkg, gvg);

  // 4) IPA core (bias, scores, softmax, all attention outputs -> cat row, bf16)
  k_attn<<<NN, 256, 0, stream>>>(z, proj, gqg, gkg, gvg, rot, trans,
                                 w_bias, head_w, catb);

  // 5) output projection: out[512,384] = cat @ w_out + b_out
  k_gemm_bf16<<<dim3(CS / 64, NN / 16), 32, 0, stream>>>(
      catb, wout_t, out, NN, CS, CATC, b_out);
}
